// Model_35338990911764
// MI455X (gfx1250) — matrix operand order repaired and checked by reading, not compiled, not run on hardware
//
#include <hip/hip_runtime.h>
#include <hip/hip_bf16.h>
#include <math.h>

// ---------------------------------------------------------------------------
// Problem constants (match reference)
// ---------------------------------------------------------------------------
#define HB   512          // hidden
#define EB   768          // embedding
#define GB   2048         // 4*H
#define BATCH 256
#define SEQ  256          // encoder steps
#define LCH  256          // chord steps (MAXM*CDL)
#define TMEL 512          // melody steps
#define CPL  9
#define MPL  38
#define CPL_PAD 16
#define MPL_PAD 48

typedef __attribute__((ext_vector_type(16))) __bf16 v16bf;
typedef __attribute__((ext_vector_type(8)))  float  v8f;

struct F16x16 { union { uint4 u[2]; v16bf v; }; };

static __device__ inline v8f wmma_bf16(v16bf a, v16bf b, v8f c) {
  // D = A(16x32 bf16) * B(32x16 bf16) + C(16x16 f32)
  return __builtin_amdgcn_wmma_f32_16x16x32_bf16(
      /*neg_a=*/false, a, /*neg_b=*/false, b,
      /*c_mod=*/(short)0, c, /*reuse_a=*/false, /*reuse_b=*/false);
}

static __device__ inline float sigm(float x) { return 1.0f / (1.0f + __expf(-x)); }

// ---------------------------------------------------------------------------
// Elementwise helpers
// ---------------------------------------------------------------------------
__global__ void cvt_f32_bf16_kernel(const float* __restrict__ s,
                                    __bf16* __restrict__ d, size_t n) {
  for (size_t i = (size_t)blockIdx.x * blockDim.x + threadIdx.x; i < n;
       i += (size_t)gridDim.x * blockDim.x)
    d[i] = (__bf16)s[i];
}

__global__ void bias_fuse_kernel(const float* __restrict__ a,
                                 const float* __restrict__ b,
                                 float* __restrict__ o, int n) {
  int i = blockIdx.x * blockDim.x + threadIdx.x;
  if (i < n) o[i] = a[i] + b[i];
}

// zero-pad small output heads (CPL/MPL) up to a 16-col multiple
__global__ void pad_w2_kernel(const float* __restrict__ W2, const float* __restrict__ b2,
                              __bf16* __restrict__ Wp, float* __restrict__ bp,
                              int Nout, int NP) {
  int idx = blockIdx.x * blockDim.x + threadIdx.x;   // NP*512 threads
  if (idx >= NP * HB) return;
  int n = idx / HB, k = idx - n * HB;
  Wp[idx] = (n < Nout) ? (__bf16)W2[n * HB + k] : (__bf16)0.0f;
  if (k == 0) bp[n] = (n < Nout) ? b2[n] : 0.0f;
}

__global__ void add_cvt_kernel(const float* __restrict__ a, const float* __restrict__ b,
                               __bf16* __restrict__ o, size_t n) {
  size_t i = (size_t)blockIdx.x * blockDim.x + threadIdx.x;
  if (i < n) o[i] = (__bf16)(a[i] + b[i]);
}

__global__ void zero1_kernel(float* p) { *p = 0.0f; }

// ---------------------------------------------------------------------------
// Generic GEMM: C[M,N] = A[M,K](bf16) * W[N,K]^T(bf16) + bias[N]
// flags: bit0 = ReLU, bit1 = bf16 output
// Each wave computes a 16x(NT*16) strip. NT is a compile-time template param
// (dispatched via scalar switch) so acc[]/b[] are statically register
// allocated -- no movrel indexed-register access, no per-iteration branches.
// Wave/tile indices are scalarized via readfirstlane so EXEC stays all-ones
// around the WMMAs (ISA requirement).
// ---------------------------------------------------------------------------
template <int NT>
static __device__ __forceinline__ void gemm_tiles(
    const __bf16* __restrict__ A, const __bf16* __restrict__ W,
    const float* __restrict__ bias, void* __restrict__ Cout,
    int N, int K, int flags, int mt, int nq, int lane) {
  const int lmod = lane & 15;
  const int kba = (lane >> 4) << 4;             // A: one 16-elem chunk (matches B)
  const int kbb = (lane >> 4) << 4;             // B: one 16-elem chunk  (k = 16*half + s, same order as A)
  const __bf16* Arow = A + (size_t)((mt << 4) + lmod) * K;

  v8f acc[NT];
#pragma unroll
  for (int t = 0; t < NT; ++t) acc[t] = v8f{0.f};
  const __bf16* Wp[NT];
#pragma unroll
  for (int t = 0; t < NT; ++t)
    Wp[t] = W + (size_t)((((nq << 2) + t) << 4) + lmod) * K + kbb;

#pragma unroll 2
  for (int k0 = 0; k0 < K; k0 += 32) {
    F16x16 a, b[NT];
    a.u[0] = *(const uint4*)(Arow + k0 + kba);
    a.u[1] = *(const uint4*)(Arow + k0 + kba + 8);
#pragma unroll
    for (int t = 0; t < NT; ++t) {
      b[t].u[0] = *(const uint4*)(Wp[t] + k0);
      b[t].u[1] = *(const uint4*)(Wp[t] + k0 + 8);
    }
    __builtin_prefetch((const void*)(Arow + k0 + 128), 0, 1);
#pragma unroll
    for (int t = 0; t < NT; ++t)
      acc[t] = wmma_bf16(a.v, b[t].v, acc[t]);
  }

  const bool relu = flags & 1, obf = flags & 2;
#pragma unroll
  for (int t = 0; t < NT; ++t) {
    const int n = (((nq << 2) + t) << 4) + lmod;
    const float bv = bias ? bias[n] : 0.0f;
#pragma unroll
    for (int v = 0; v < 8; ++v) {
      const int m = (mt << 4) + ((lane >> 4) << 3) + v;   // ISA C/D layout
      float x = acc[t][v] + bv;
      if (relu && x < 0.0f) x = 0.0f;
      const size_t o = (size_t)m * N + n;
      if (obf) ((__bf16*)Cout)[o] = (__bf16)x;
      else     ((float*)Cout)[o] = x;
    }
  }
}

__global__ __launch_bounds__(256) void gemm_bf16_kernel(
    const __bf16* __restrict__ A, const __bf16* __restrict__ W,
    const float* __restrict__ bias, void* __restrict__ Cout,
    int M, int N, int K, int flags) {
  const int tilesN4 = (N + 63) >> 6;
  const int wavesTotal = (M >> 4) * tilesN4;
  const int wib = __builtin_amdgcn_readfirstlane((int)threadIdx.x >> 5);  // scalar
  const int gw = (int)blockIdx.x * 8 + wib;                               // scalar
  if (gw >= wavesTotal) return;                 // scalar branch
  const int mt = gw / tilesN4;                  // scalar
  const int nq = gw - mt * tilesN4;             // scalar
  const int lane = threadIdx.x & 31;
  int nTiles = (N >> 4) - (nq << 2);            // scalar
  if (nTiles > 4) nTiles = 4;
  switch (nTiles) {                             // scalar dispatch
    case 4: gemm_tiles<4>(A, W, bias, Cout, N, K, flags, mt, nq, lane); break;
    case 3: gemm_tiles<3>(A, W, bias, Cout, N, K, flags, mt, nq, lane); break;
    case 2: gemm_tiles<2>(A, W, bias, Cout, N, K, flags, mt, nq, lane); break;
    default: gemm_tiles<1>(A, W, bias, Cout, N, K, flags, mt, nq, lane); break;
  }
}

// ---------------------------------------------------------------------------
// LSTM recurrence. 16 WGs x 16 batch rows, 512 threads (16 waves).
// Gin[row=(b*T+t), 4H] bf16 already holds x@Wih^T + bih + bhh.
// Wave w owns interleaved columns {g*512 + w*32 .. +31 | g in i,f,g,o} so the
// gate nonlinearity runs on its own WMMA accumulators; h lives in LDS.
// Each step the 64KB gate-input slab for timestep t is pulled into LDS with
// CDNA5 async global->LDS loads (ASYNCcnt), overlapping the recurrent GEMM.
// ---------------------------------------------------------------------------
__global__ __launch_bounds__(512) void lstm_rec_kernel(
    const __bf16* __restrict__ Gin, const __bf16* __restrict__ Whh,
    const float* __restrict__ h0, const float* __restrict__ c0,
    __bf16* __restrict__ hs_out, float* __restrict__ hfinal,
    int T, int reverse) {
  __shared__ __attribute__((aligned(16))) __bf16 sh[16 * HB];    // 16 KB: h
  __shared__ __attribute__((aligned(16))) __bf16 sgin[16 * GB];  // 64 KB: gate inputs

  const int tid = threadIdx.x;
  const int w = __builtin_amdgcn_readfirstlane(tid >> 5);        // scalar wave id
  const int lane = tid & 31;
  const int lmod = lane & 15;
  const int b0 = blockIdx.x << 4;
  const int mbase = (lane >> 4) << 3;
  const int colb = (w << 5) + lmod;             // wave-local column base (0..511)
  const int kba = (lane >> 4) << 4;
  const int kbb = (lane >> 4) << 4;

  // init h (LDS) and c (registers)
  {
    const int m = tid >> 5, kk0 = (tid & 31) << 4;
    for (int i = 0; i < 16; ++i) {
      float hv = h0 ? h0[(size_t)(b0 + m) * HB + kk0 + i] : 0.0f;
      sh[m * HB + kk0 + i] = (__bf16)hv;
    }
  }
  float creg[2][8], hreg[2][8];
#pragma unroll
  for (int s = 0; s < 2; ++s)
#pragma unroll
    for (int v = 0; v < 8; ++v)
      creg[s][v] = c0 ? c0[(size_t)(b0 + mbase + v) * HB + colb + (s << 4)] : 0.0f;
  __syncthreads();

  const __bf16* shrow = sh + (size_t)lmod * HB;
  const char* gbase = (const char*)Gin;
  char* sgb = (char*)sgin;

  // hoisted per-wave Whh base pointers for the 8 owned column tiles
  const __bf16* Wb[8];
#pragma unroll
  for (int nt = 0; nt < 8; ++nt) {
    const int gate = nt >> 1, sub = nt & 1;
    const int n = gate * HB + (w << 5) + (sub << 4) + lmod;
    Wb[nt] = Whh + (size_t)n * HB + kbb;
  }
  // hoisted per-thread async-copy offsets (byte offsets, 8 chunks of 16B)
  size_t rowOff[8];
  unsigned ldsOff[8];
#pragma unroll
  for (int j = 0; j < 8; ++j) {
    const int f = tid * 128 + j * 16;            // byte offset in 64KB slab
    const int m = f >> 12;                       // row 0..15
    const int inrow = f & 4095;                  // byte within 4KB row
    rowOff[j] = (size_t)(b0 + m) * T * (size_t)(GB * 2) + (size_t)inrow;
    ldsOff[j] = (unsigned)(uintptr_t)(sgb + f);
  }

  for (int t = 0; t < T; ++t) {
    const int tA = reverse ? (T - 1 - t) : t;

    // ---- issue async copy of this step's gate inputs (16 rows x 4 KB -> LDS)
    const size_t tOff = (size_t)tA * (size_t)(GB * 2);
#pragma unroll
    for (int j = 0; j < 8; ++j) {
      const unsigned long long gaddr =
          (unsigned long long)(uintptr_t)(gbase + rowOff[j] + tOff);
      asm volatile("global_load_async_to_lds_b128 %0, %1, off"
                   :: "v"(ldsOff[j]), "v"(gaddr) : "memory");
    }

    // ---- recurrent GEMM h@Whh^T (reads old h from LDS)
    v8f acc[8];
#pragma unroll
    for (int nt = 0; nt < 8; ++nt) acc[nt] = v8f{0.f};
#pragma unroll 2
    for (int k0 = 0; k0 < HB; k0 += 32) {
      F16x16 a, b4[4];
      a.u[0] = *(const uint4*)(shrow + k0 + kba);
      a.u[1] = *(const uint4*)(shrow + k0 + kba + 8);
#pragma unroll
      for (int nt = 0; nt < 4; ++nt) {
        b4[nt].u[0] = *(const uint4*)(Wb[nt] + k0);
        b4[nt].u[1] = *(const uint4*)(Wb[nt] + k0 + 8);
      }
#pragma unroll
      for (int nt = 0; nt < 4; ++nt)
        acc[nt] = wmma_bf16(a.v, b4[nt].v, acc[nt]);
#pragma unroll
      for (int nt = 0; nt < 4; ++nt) {
        b4[nt].u[0] = *(const uint4*)(Wb[4 + nt] + k0);
        b4[nt].u[1] = *(const uint4*)(Wb[4 + nt] + k0 + 8);
      }
#pragma unroll
      for (int nt = 0; nt < 4; ++nt)
        acc[4 + nt] = wmma_bf16(a.v, b4[nt].v, acc[4 + nt]);
    }

    // ---- wait for my async copies; barrier covers everyone's copies AND
    //      everyone's reads of old h
    asm volatile("s_wait_asynccnt 0" ::: "memory");
    __syncthreads();

    // ---- gate nonlinearity (gates for my columns live in my accumulators)
#pragma unroll
    for (int s = 0; s < 2; ++s) {
      const int col = colb + (s << 4);
#pragma unroll
      for (int v = 0; v < 8; ++v) {
        const int mrow = mbase + v;
        const int gi0 = mrow * GB + col;
        const float gi = acc[s][v]     + (float)sgin[gi0];
        const float gf = acc[2 + s][v] + (float)sgin[gi0 + HB];
        const float gg = acc[4 + s][v] + (float)sgin[gi0 + 2 * HB];
        const float go = acc[6 + s][v] + (float)sgin[gi0 + 3 * HB];
        const float cc = sigm(gf) * creg[s][v] + sigm(gi) * tanhf(gg);
        creg[s][v] = cc;
        const float hh = sigm(go) * tanhf(cc);
        hreg[s][v] = hh;
        const __bf16 hb = (__bf16)hh;
        sh[mrow * HB + col] = hb;
        if (hs_out)
          hs_out[((size_t)(b0 + mrow) * T + tA) * HB + col] = hb;
      }
    }
    __syncthreads();                 // new h visible before next step's reads
  }
  if (hfinal) {
#pragma unroll
    for (int s = 0; s < 2; ++s)
#pragma unroll
      for (int v = 0; v < 8; ++v)
        hfinal[(size_t)(b0 + mbase + v) * HB + colb + (s << 4)] = hreg[s][v];
  }
}

// ---------------------------------------------------------------------------
// VAE head helpers
// ---------------------------------------------------------------------------
__global__ __launch_bounds__(256) void z_kl_kernel(
    const float* __restrict__ mu, const float* __restrict__ lv,
    const float* __restrict__ eps, __bf16* __restrict__ zbf,
    float* __restrict__ klout) {
  __shared__ float red[256];
  const size_t idx = (size_t)blockIdx.x * 256 + threadIdx.x;   // BATCH*HB total
  const float m = mu[idx], l = lv[idx];
  zbf[idx] = (__bf16)(m + eps[idx] * __expf(0.5f * l));
  red[threadIdx.x] = 0.5f * (m * m + __expf(l) - l - 1.0f) * (1.0f / (BATCH * HB));
  __syncthreads();
  for (int off = 128; off; off >>= 1) {
    if ((int)threadIdx.x < off) red[threadIdx.x] += red[threadIdx.x + off];
    __syncthreads();
  }
  if (threadIdx.x == 0) atomicAdd(klout, red[0]);
}

// out[b] = dot(Hid[b,:512], w) + b0   (N=1 head)
__global__ __launch_bounds__(256) void dot_head_kernel(
    const float* __restrict__ Hid, const float* __restrict__ w,
    const float* __restrict__ b, float* __restrict__ out) {
  __shared__ float red[256];
  const int bi = blockIdx.x, t = threadIdx.x;
  red[t] = Hid[(size_t)bi * HB + t] * w[t] + Hid[(size_t)bi * HB + 256 + t] * w[256 + t];
  __syncthreads();
  for (int off = 128; off; off >>= 1) {
    if (t < off) red[t] += red[t + off];
    __syncthreads();
  }
  if (t == 0) out[bi] = red[0] + b[0];
}

// zcat[b, 0:512]=z, [512:1024]=bpm*W+b, [1024:1536]=val*W+b, [1536:2048]=en*W+b
__global__ void build_zcat_kernel(const __bf16* __restrict__ zbf,
                                  const float* __restrict__ bpm, const float* __restrict__ val,
                                  const float* __restrict__ en,
                                  const float* __restrict__ bW, const float* __restrict__ bB,
                                  const float* __restrict__ vW, const float* __restrict__ vB,
                                  const float* __restrict__ eW, const float* __restrict__ eB,
                                  __bf16* __restrict__ zcat) {
  const int idx = blockIdx.x * 256 + threadIdx.x;    // BATCH*GB
  const int b = idx >> 11, j = idx & 2047;
  float v;
  if (j < 512)       v = (float)zbf[b * HB + j];
  else if (j < 1024) { int k = j - 512;  v = bpm[b] * bW[k] + bB[k]; }
  else if (j < 1536) { int k = j - 1024; v = val[b] * vW[k] + vB[k]; }
  else               { int k = j - 1536; v = en[b]  * eW[k] + eB[k]; }
  zcat[idx] = (__bf16)v;
}

// chord LSTM inputs: t==0 -> z2, else chord_table[gt_chords[:,t-1]]
__global__ void build_xch_kernel(const float* __restrict__ z2, const int* __restrict__ gt,
                                 const float* __restrict__ table, __bf16* __restrict__ X,
                                 size_t n) {
  for (size_t idx = (size_t)blockIdx.x * blockDim.x + threadIdx.x; idx < n;
       idx += (size_t)gridDim.x * blockDim.x) {
    const int k = idx & 511;
    const size_t r = idx >> 9;
    const int t = (int)(r & (LCH - 1)), b = (int)(r >> 8);
    const float v = (t == 0) ? z2[b * HB + k]
                             : table[(size_t)gt[b * LCH + t - 1] * HB + k];
    X[idx] = (__bf16)v;
  }
}

// melody LSTM inputs: chord_emb[k/2] + (k>0 ? mel_table[gt_melody[:,k-1]] : 0)
__global__ void build_xmel_kernel(const int* __restrict__ gtc, const int* __restrict__ gtm,
                                  const float* __restrict__ ctab, const float* __restrict__ mtab,
                                  __bf16* __restrict__ X, size_t n) {
  for (size_t idx = (size_t)blockIdx.x * blockDim.x + threadIdx.x; idx < n;
       idx += (size_t)gridDim.x * blockDim.x) {
    const int k = idx & 511;
    const size_t r = idx >> 9;
    const int kt = (int)(r & (TMEL - 1)), b = (int)(r >> 9);
    float v = ctab[(size_t)gtc[b * LCH + (kt >> 1)] * HB + k];
    if (kt > 0) v += mtab[(size_t)gtm[b * (TMEL - 1) + kt - 1] * HB + k];
    X[idx] = (__bf16)v;
  }
}

__global__ void extract_kernel(const float* __restrict__ pad, float* __restrict__ out,
                               int NP, int Nout, size_t n) {
  for (size_t idx = (size_t)blockIdx.x * blockDim.x + threadIdx.x; idx < n;
       idx += (size_t)gridDim.x * blockDim.x) {
    const int c = (int)(idx % Nout);
    const size_t rt = idx / Nout;                 // = b*T + t (b-major, matches GEMM rows)
    out[idx] = pad[rt * NP + c];
  }
}

// ---------------------------------------------------------------------------
// Host orchestration
// ---------------------------------------------------------------------------
extern "C" void kernel_launch(void* const* d_in, const int* in_sizes, int n_in,
                              void* d_out, int out_size, void* d_ws, size_t ws_size,
                              hipStream_t stream) {
  (void)in_sizes; (void)n_in; (void)out_size; (void)ws_size;
  // ---- inputs
  const float* x_in   = (const float*)d_in[0];
  const float* eps    = (const float*)d_in[1];
  const float* hx_ch  = (const float*)d_in[2];
  const float* cx_ch  = (const float*)d_in[3];
  const float* hx_me  = (const float*)d_in[4];
  const float* cx_me  = (const float*)d_in[5];
  const int*   gt_ch  = (const int*)d_in[6];
  const int*   gt_me  = (const int*)d_in[7];
#define PRM(i) ((const float*)d_in[9 + (i)])
  const float *enc_Wih_f=PRM(0), *enc_Whh_f=PRM(1), *enc_bih_f=PRM(2), *enc_bhh_f=PRM(3);
  const float *enc_Wih_b=PRM(4), *enc_Whh_b=PRM(5), *enc_bih_b=PRM(6), *enc_bhh_b=PRM(7);
  const float *mean_W=PRM(8), *mean_b=PRM(9), *var_W=PRM(10), *var_b=PRM(11);
  const float *down_W=PRM(12), *down_b=PRM(13);
  const float *bpm1_W=PRM(14), *bpm1_b=PRM(15), *bpm2_W=PRM(16), *bpm2_b=PRM(17);
  const float *val1_W=PRM(18), *val1_b=PRM(19), *val2_W=PRM(20), *val2_b=PRM(21);
  const float *en1_W=PRM(22),  *en1_b=PRM(23),  *en2_W=PRM(24),  *en2_b=PRM(25);
  const float *bpm_eW=PRM(26), *bpm_eB=PRM(27), *val_eW=PRM(28), *val_eB=PRM(29);
  const float *en_eW=PRM(30),  *en_eB=PRM(31);
  const float *ch_Wih=PRM(32), *ch_Whh=PRM(33), *ch_bih=PRM(34), *ch_bhh=PRM(35);
  const float *me_Wih=PRM(36), *me_Whh=PRM(37), *me_bih=PRM(38), *me_bhh=PRM(39);
  const float *chord_table=PRM(40), *mel_table=PRM(41);
  const float *chl1_W=PRM(42), *chl1_b=PRM(43), *chl2_W=PRM(44), *chl2_b=PRM(45);
  const float *ml1_W=PRM(46),  *ml1_b=PRM(47),  *ml2_W=PRM(48),  *ml2_b=PRM(49);

  // ---- outputs (flat f32, tuple order)
  float* out = (float*)d_out;
  float* outCh  = out;                                           // [B,L,CPL]
  float* outMel = out + (size_t)BATCH * LCH * CPL;               // [B,T,MPL]
  float* outBpm = outMel + (size_t)BATCH * TMEL * MPL;           // [B,1]
  float* outVal = outBpm + BATCH;
  float* outEn  = outVal + BATCH;
  float* outKl  = outEn + BATCH;                                 // scalar

  // ---- workspace bump allocator
  char* ws = (char*)d_ws;
  size_t cur = 0;
  auto alloc = [&](size_t bytes) -> char* {
    cur = (cur + 255) & ~(size_t)255;
    char* p = ws + cur; cur += bytes; return p;
  };
  __bf16* wEncIhF = (__bf16*)alloc((size_t)GB * EB * 2);
  __bf16* wEncHhF = (__bf16*)alloc((size_t)GB * HB * 2);
  __bf16* wEncIhB = (__bf16*)alloc((size_t)GB * EB * 2);
  __bf16* wEncHhB = (__bf16*)alloc((size_t)GB * HB * 2);
  __bf16* wMean   = (__bf16*)alloc((size_t)HB * HB * 2);
  __bf16* wVar    = (__bf16*)alloc((size_t)HB * HB * 2);
  __bf16* wDown   = (__bf16*)alloc((size_t)HB * GB * 2);
  __bf16* wBpm1   = (__bf16*)alloc((size_t)HB * HB * 2);
  __bf16* wVal1   = (__bf16*)alloc((size_t)HB * HB * 2);
  __bf16* wEn1    = (__bf16*)alloc((size_t)HB * HB * 2);
  __bf16* wChIh   = (__bf16*)alloc((size_t)GB * HB * 2);
  __bf16* wChHh   = (__bf16*)alloc((size_t)GB * HB * 2);
  __bf16* wMeIh   = (__bf16*)alloc((size_t)GB * HB * 2);
  __bf16* wMeHh   = (__bf16*)alloc((size_t)GB * HB * 2);
  __bf16* wChl1   = (__bf16*)alloc((size_t)HB * HB * 2);
  __bf16* wMl1    = (__bf16*)alloc((size_t)HB * HB * 2);
  __bf16* wCh2p   = (__bf16*)alloc((size_t)CPL_PAD * HB * 2);
  __bf16* wMl2p   = (__bf16*)alloc((size_t)MPL_PAD * HB * 2);
  float* bEncF = (float*)alloc(GB * 4);
  float* bEncB = (float*)alloc(GB * 4);
  float* bCh   = (float*)alloc(GB * 4);
  float* bMe   = (float*)alloc(GB * 4);
  float* bCh2p = (float*)alloc(CPL_PAD * 4);
  float* bMl2p = (float*)alloc(MPL_PAD * 4);
  float*  hf     = (float*)alloc((size_t)BATCH * HB * 4);
  float*  hb     = (float*)alloc((size_t)BATCH * HB * 4);
  __bf16* tempbf = (__bf16*)alloc((size_t)BATCH * HB * 2);
  float*  muB    = (float*)alloc((size_t)BATCH * HB * 4);
  float*  lvB    = (float*)alloc((size_t)BATCH * HB * 4);
  __bf16* zbf    = (__bf16*)alloc((size_t)BATCH * HB * 2);
  float*  hidden = (float*)alloc((size_t)BATCH * HB * 4);
  __bf16* zcat   = (__bf16*)alloc((size_t)BATCH * GB * 2);
  float*  z2f    = (float*)alloc((size_t)BATCH * HB * 4);
  float*  padAr = (float*)alloc((size_t)BATCH * TMEL * MPL_PAD * 4);  // 25 MB
  __bf16* xAr   = (__bf16*)alloc((size_t)BATCH * SEQ * EB * 2);       // 100.7 MB arena
  __bf16* hsAr  = (__bf16*)alloc((size_t)BATCH * TMEL * HB * 2);      // 128 MB
  __bf16* hidAr = (__bf16*)alloc((size_t)BATCH * TMEL * HB * 2);      // 128 MB
  __bf16* ginAr = (__bf16*)alloc((size_t)BATCH * TMEL * GB * 2);      // 536 MB

  auto cvt = [&](const float* s, __bf16* d, size_t n) {
    int blocks = (int)((n + 256 * 8 - 1) / (256 * 8)); if (blocks > 16384) blocks = 16384;
    cvt_f32_bf16_kernel<<<blocks, 256, 0, stream>>>(s, d, n);
  };
  auto gemm = [&](const __bf16* A, const __bf16* W, const float* bias, void* C,
                  int M, int N, int K, int flags) {
    int tiles = (N + 63) / 64;
    int waves = (M / 16) * tiles;
    gemm_bf16_kernel<<<(waves + 7) / 8, 256, 0, stream>>>(A, W, bias, C, M, N, K, flags);
  };

  // ---- 0) convert weights, fuse biases, pad small heads
  cvt(enc_Wih_f, wEncIhF, (size_t)GB * EB);  cvt(enc_Whh_f, wEncHhF, (size_t)GB * HB);
  cvt(enc_Wih_b, wEncIhB, (size_t)GB * EB);  cvt(enc_Whh_b, wEncHhB, (size_t)GB * HB);
  cvt(mean_W, wMean, (size_t)HB * HB);       cvt(var_W, wVar, (size_t)HB * HB);
  cvt(down_W, wDown, (size_t)HB * GB);
  cvt(bpm1_W, wBpm1, (size_t)HB * HB);       cvt(val1_W, wVal1, (size_t)HB * HB);
  cvt(en1_W, wEn1, (size_t)HB * HB);
  cvt(ch_Wih, wChIh, (size_t)GB * HB);       cvt(ch_Whh, wChHh, (size_t)GB * HB);
  cvt(me_Wih, wMeIh, (size_t)GB * HB);       cvt(me_Whh, wMeHh, (size_t)GB * HB);
  cvt(chl1_W, wChl1, (size_t)HB * HB);       cvt(ml1_W, wMl1, (size_t)HB * HB);
  bias_fuse_kernel<<<GB / 256, 256, 0, stream>>>(enc_bih_f, enc_bhh_f, bEncF, GB);
  bias_fuse_kernel<<<GB / 256, 256, 0, stream>>>(enc_bih_b, enc_bhh_b, bEncB, GB);
  bias_fuse_kernel<<<GB / 256, 256, 0, stream>>>(ch_bih, ch_bhh, bCh, GB);
  bias_fuse_kernel<<<GB / 256, 256, 0, stream>>>(me_bih, me_bhh, bMe, GB);
  pad_w2_kernel<<<(CPL_PAD * HB + 255) / 256, 256, 0, stream>>>(chl2_W, chl2_b, wCh2p, bCh2p, CPL, CPL_PAD);
  pad_w2_kernel<<<(MPL_PAD * HB + 255) / 256, 256, 0, stream>>>(ml2_W, ml2_b, wMl2p, bMl2p, MPL, MPL_PAD);

  // ---- 1) encoder
  cvt(x_in, xAr, (size_t)BATCH * SEQ * EB);                                 // X bf16 (rows b*S+t)
  gemm(xAr, wEncIhF, bEncF, ginAr, BATCH * SEQ, GB, EB, /*bf16*/2);         // gate inputs fwd
  lstm_rec_kernel<<<BATCH / 16, 512, 0, stream>>>(ginAr, wEncHhF, nullptr, nullptr,
                                                  nullptr, hf, SEQ, 0);
  gemm(xAr, wEncIhB, bEncB, ginAr, BATCH * SEQ, GB, EB, 2);                 // gate inputs bwd
  lstm_rec_kernel<<<BATCH / 16, 512, 0, stream>>>(ginAr, wEncHhB, nullptr, nullptr,
                                                  nullptr, hb, SEQ, 1);
  add_cvt_kernel<<<(BATCH * HB) / 256, 256, 0, stream>>>(hf, hb, tempbf, (size_t)BATCH * HB);
  gemm(tempbf, wMean, mean_b, muB, BATCH, HB, HB, 0);
  gemm(tempbf, wVar,  var_b,  lvB, BATCH, HB, HB, 0);
  zero1_kernel<<<1, 1, 0, stream>>>(outKl);
  z_kl_kernel<<<(BATCH * HB) / 256, 256, 0, stream>>>(muB, lvB, eps, zbf, outKl);

  // ---- 2) heads
  gemm(zbf, wBpm1, bpm1_b, hidden, BATCH, HB, HB, /*relu*/1);
  dot_head_kernel<<<BATCH, 256, 0, stream>>>(hidden, bpm2_W, bpm2_b, outBpm);
  gemm(zbf, wVal1, val1_b, hidden, BATCH, HB, HB, 1);
  dot_head_kernel<<<BATCH, 256, 0, stream>>>(hidden, val2_W, val2_b, outVal);
  gemm(zbf, wEn1, en1_b, hidden, BATCH, HB, HB, 1);
  dot_head_kernel<<<BATCH, 256, 0, stream>>>(hidden, en2_W, en2_b, outEn);
  build_zcat_kernel<<<(BATCH * GB) / 256, 256, 0, stream>>>(
      zbf, outBpm, outVal, outEn, bpm_eW, bpm_eB, val_eW, val_eB, en_eW, en_eB, zcat);
  gemm(zcat, wDown, down_b, z2f, BATCH, HB, GB, 0);

  // ---- 3) chord LSTM
  build_xch_kernel<<<16384, 256, 0, stream>>>(z2f, gt_ch, chord_table, xAr,
                                              (size_t)BATCH * LCH * HB);
  gemm(xAr, wChIh, bCh, ginAr, BATCH * LCH, GB, HB, 2);
  lstm_rec_kernel<<<BATCH / 16, 512, 0, stream>>>(ginAr, wChHh, hx_ch, cx_ch,
                                                  hsAr, nullptr, LCH, 0);
  gemm(hsAr, wChl1, chl1_b, hidAr, BATCH * LCH, HB, HB, /*relu|bf16*/3);
  gemm(hidAr, wCh2p, bCh2p, padAr, BATCH * LCH, CPL_PAD, HB, 0);
  extract_kernel<<<16384, 256, 0, stream>>>(padAr, outCh, CPL_PAD, CPL,
                                            (size_t)BATCH * LCH * CPL);

  // ---- 4) melody LSTM
  build_xmel_kernel<<<16384, 256, 0, stream>>>(gt_ch, gt_me, chord_table, mel_table,
                                               xAr, (size_t)BATCH * TMEL * HB);
  gemm(xAr, wMeIh, bMe, ginAr, BATCH * TMEL, GB, HB, 2);
  lstm_rec_kernel<<<BATCH / 16, 512, 0, stream>>>(ginAr, wMeHh, hx_me, cx_me,
                                                  hsAr, nullptr, TMEL, 0);
  gemm(hsAr, wMl1, ml1_b, hidAr, BATCH * TMEL, HB, HB, 3);
  gemm(hidAr, wMl2p, bMl2p, padAr, BATCH * TMEL, MPL_PAD, HB, 0);
  extract_kernel<<<16384, 256, 0, stream>>>(padAr, outMel, MPL_PAD, MPL,
                                            (size_t)BATCH * TMEL * MPL);
#undef PRM
}
